// GGMLLayer_37666863186343
// MI455X (gfx1250) — matrix operand order repaired and checked by reading, not compiled, not run on hardware
//
#include <hip/hip_runtime.h>

typedef _Float16 v16h __attribute__((ext_vector_type(16)));
typedef _Float16 v8h  __attribute__((ext_vector_type(8)));
typedef _Float16 v4h  __attribute__((ext_vector_type(4)));
typedef float    v8f  __attribute__((ext_vector_type(8)));
typedef float    v4f  __attribute__((ext_vector_type(4)));
typedef int      v4i  __attribute__((ext_vector_type(4)));

// Problem dims fixed by the reference.
constexpr int T = 2048;   // tokens
constexpr int I = 4096;   // in features (K)
constexpr int O = 4096;   // out features (N)
constexpr int R = 16;     // LoRA rank
constexpr int QB = 32;    // quant block

// GEMM tiling
constexpr int BM = 128;         // tokens per block
constexpr int BN = 128;         // out-features per block
constexpr int BK = 32;          // K-step (== quant block -> one scale/row/step)
constexpr int LDSS = 40;        // LDS row stride in halves (80B: 16B-aligned, bank de-phased)
constexpr int KSTEPS = I / BK;  // 128

// Per-thread staging registers for one (A,B) tile fetch (double buffering).
struct Stage {
    v4f   xr[4];
    v4i   qr[4];
    float sr[4];
};

__device__ __forceinline__ void fetch_tile(
    const float* __restrict__ x, const int* __restrict__ q,
    const float* __restrict__ scales,
    int block_m, int block_n, int k0, int tid, Stage& st)
{
    #pragma unroll
    for (int it = 0; it < 4; ++it) {
        const int e = (it * 256 + tid) * 4;      // 0..4095
        const int r = e >> 5;
        const int c = e & 31;
        st.xr[it] = *(const v4f*)(x + (size_t)(block_m + r) * I + k0 + c);
    }
    #pragma unroll
    for (int it = 0; it < 4; ++it) {
        const int e = (it * 256 + tid) * 4;
        const int r = e >> 5;
        const int c = e & 31;
        const int orow = block_n + r;
        st.qr[it] = *(const v4i*)(q + (size_t)orow * I + k0 + c);
        st.sr[it] = scales[(size_t)orow * (I / QB) + (k0 >> 5)];
    }
}

__device__ __forceinline__ void store_tile(
    _Float16* __restrict__ Al, _Float16* __restrict__ Bl, int tid, const Stage& st)
{
    #pragma unroll
    for (int it = 0; it < 4; ++it) {
        const int e = (it * 256 + tid) * 4;
        const int r = e >> 5;
        const int c = e & 31;
        v4h h;
        h[0] = (_Float16)st.xr[it][0]; h[1] = (_Float16)st.xr[it][1];
        h[2] = (_Float16)st.xr[it][2]; h[3] = (_Float16)st.xr[it][3];
        *(v4h*)(&Al[r * LDSS + c]) = h;
        v4h g;
        g[0] = (_Float16)((float)(st.qr[it][0] - 8) * st.sr[it]);
        g[1] = (_Float16)((float)(st.qr[it][1] - 8) * st.sr[it]);
        g[2] = (_Float16)((float)(st.qr[it][2] - 8) * st.sr[it]);
        g[3] = (_Float16)((float)(st.qr[it][3] - 8) * st.sr[it]);
        *(v4h*)(&Bl[r * LDSS + c]) = g;
    }
}

// ---------------------------------------------------------------------------
// Kernel 1: y = x @ dequant(q,scales)^T + bias   via v_wmma_f32_16x16x32_f16
// 256 threads = 8 waves; wave grid 2(M)x4(N); wave tile 64x32 = 4x2 WMMA.
// Double-buffered LDS: global fetch of tile k+1 overlaps WMMA on tile k.
// ---------------------------------------------------------------------------
__global__ __launch_bounds__(256) void qgemm_wmma(
    const float* __restrict__ x, const int* __restrict__ q,
    const float* __restrict__ scales, const float* __restrict__ bias,
    float* __restrict__ y)
{
    __shared__ __align__(16) _Float16 Al[2][BM * LDSS];   // x tile, [row][k]
    __shared__ __align__(16) _Float16 Bl[2][BN * LDSS];   // W tile, [o][k]

    const int tid  = threadIdx.x;
    const int wave = tid >> 5;
    const int lane = tid & 31;
    const int wm   = wave & 1;        // 0..1  (M)
    const int wn   = wave >> 1;       // 0..3  (N)
    const int block_m = blockIdx.y * BM;
    const int block_n = blockIdx.x * BN;

    const int lrow  = lane & 15;      // row/col within 16
    const int lhalf = lane & 16;      // upper/lower half-wave

    v8f acc[4][2] = {};

    Stage st;
    fetch_tile(x, q, scales, block_m, block_n, 0, tid, st);
    store_tile(&Al[0][0], &Bl[0][0], tid, st);
    __syncthreads();

    for (int kb = 0; kb < KSTEPS; ++kb) {
        const int  cur  = kb & 1;
        const bool more = (kb + 1) < KSTEPS;

        // Prefetch next tile into registers; latency hides under the WMMAs.
        if (more)
            fetch_tile(x, q, scales, block_m, block_n, (kb + 1) * BK, tid, st);

        // ---- A fragments: ISA 16-bit A 16x32 layout ----
        // lanes 0-15: M=lane, VGPR0-3 = K0..7,  VGPR4-7 = K16..23
        // lanes16-31: M=lane-16,       K8..15,          K24..31
        v16h afrag[4];
        #pragma unroll
        for (int mt = 0; mt < 4; ++mt) {
            const int row = wm * 64 + mt * 16 + lrow;
            const int klo = lhalf ? 8 : 0;
            const v8h a0 = *(const v8h*)(&Al[cur][row * LDSS + klo]);
            const v8h a1 = *(const v8h*)(&Al[cur][row * LDSS + 16 + klo]);
            #pragma unroll
            for (int i = 0; i < 8; ++i) { afrag[mt][i] = a0[i]; afrag[mt][i + 8] = a1[i]; }
        }
        // ---- B fragments: 32x16 K-major; lanes 0-15 hold K0..7,K16..23; 16-31 hold K8..15,K24..31 ----
        v16h bfrag[2];
        #pragma unroll
        for (int nt = 0; nt < 2; ++nt) {
            const int row = wn * 32 + nt * 16 + lrow;    // output feature (column of B)
            const int kb2 = lhalf ? 8 : 0;               // 0 or 8
            const v8h b0 = *(const v8h*)(&Bl[cur][row * LDSS + kb2]);
            const v8h b1 = *(const v8h*)(&Bl[cur][row * LDSS + kb2 + 16]);
            #pragma unroll
            for (int i = 0; i < 8; ++i) { bfrag[nt][i] = b0[i]; bfrag[nt][i + 8] = b1[i]; }
        }

        #pragma unroll
        for (int mt = 0; mt < 4; ++mt)
            #pragma unroll
            for (int nt = 0; nt < 2; ++nt)
                acc[mt][nt] = __builtin_amdgcn_wmma_f32_16x16x32_f16(
                    false, afrag[mt], false, bfrag[nt],
                    (short)0, acc[mt][nt], false, false);

        // Convert + store next tile into the alternate buffer (no read conflict).
        if (more)
            store_tile(&Al[cur ^ 1][0], &Bl[cur ^ 1][0], tid, st);

        __syncthreads();   // next-tile stores visible; this-tile reads done before overwrite
    }

    // ---- epilogue: C/D layout -> lane holds col = lane%16; VGPR j -> row j (+8 upper half) ----
    #pragma unroll
    for (int mt = 0; mt < 4; ++mt) {
        #pragma unroll
        for (int nt = 0; nt < 2; ++nt) {
            const int col = block_n + wn * 32 + nt * 16 + lrow;
            const float bv = bias[col];
            #pragma unroll
            for (int j = 0; j < 8; ++j) {
                const int row = block_m + wm * 64 + mt * 16 + (lhalf ? 8 : 0) + j;
                y[(size_t)row * O + col] = acc[mt][nt][j] + bv;
            }
        }
    }
}

// ---------------------------------------------------------------------------
// Kernel 2: tmp[t][r] = sum_i x[t][i] * down[r][i]   (T x R, K = I)
// Block handles 16 tokens x 16 ranks; x/down staged through LDS in 128-chunks.
// ---------------------------------------------------------------------------
__global__ __launch_bounds__(256) void lora_down_k(
    const float* __restrict__ x, const float* __restrict__ down,
    float* __restrict__ tmp)
{
    __shared__ float xs[16 * 128];
    __shared__ float ds[16 * 128];
    const int tid = threadIdx.x;
    const int tb  = blockIdx.x * 16;
    const int tl  = tid >> 4;      // token 0..15
    const int r   = tid & 15;      // rank 0..15
    float sum = 0.f;
    for (int i0 = 0; i0 < I; i0 += 128) {
        #pragma unroll
        for (int it = 0; it < 8; ++it) {
            const int e  = it * 256 + tid;   // 0..2047
            const int rr = e >> 7;
            const int cc = e & 127;
            xs[e] = x[(size_t)(tb + rr) * I + i0 + cc];
            ds[e] = down[(size_t)rr * I + i0 + cc];
        }
        __syncthreads();
        #pragma unroll 8
        for (int c = 0; c < 128; ++c)
            sum += xs[tl * 128 + c] * ds[r * 128 + c];
        __syncthreads();
    }
    tmp[(size_t)(tb + tl) * R + r] = sum;
}

// ---------------------------------------------------------------------------
// Kernel 3: y[t][o] += alpha * sum_r tmp[t][r] * up[o][r]
// ---------------------------------------------------------------------------
__global__ __launch_bounds__(256) void lora_up_k(
    const float* __restrict__ tmp, const float* __restrict__ up,
    const float* __restrict__ alpha, float* __restrict__ y)
{
    const size_t idx = (size_t)blockIdx.x * 256 + threadIdx.x;
    const int t = (int)(idx >> 12);     // / 4096
    const int o = (int)(idx & 4095);
    const v4f* tp  = (const v4f*)(tmp + (size_t)t * R);
    const v4f* upp = (const v4f*)(up  + (size_t)o * R);
    float s = 0.f;
    #pragma unroll
    for (int j = 0; j < 4; ++j) {
        const v4f a = tp[j], b = upp[j];
        s += a[0] * b[0] + a[1] * b[1] + a[2] * b[2] + a[3] * b[3];
    }
    y[idx] += alpha[0] * s;
}

extern "C" void kernel_launch(void* const* d_in, const int* in_sizes, int n_in,
                              void* d_out, int out_size, void* d_ws, size_t ws_size,
                              hipStream_t stream) {
    const float* x         = (const float*)d_in[0];
    const int*   q         = (const int*)  d_in[1];
    const float* scales    = (const float*)d_in[2];
    const float* lora_up   = (const float*)d_in[3];
    const float* lora_down = (const float*)d_in[4];
    const float* alpha     = (const float*)d_in[5];
    const float* bias      = (const float*)d_in[6];
    float* y   = (float*)d_out;
    float* tmp = (float*)d_ws;          // T*R floats = 128 KB

    dim3 g1(O / BN, T / BM);            // 32 x 16 = 512 blocks
    qgemm_wmma<<<g1, 256, 0, stream>>>(x, q, scales, bias, y);
    lora_down_k<<<T / 16, 256, 0, stream>>>(x, lora_down, tmp);
    lora_up_k<<<(int)(((size_t)T * O) / 256), 256, 0, stream>>>(tmp, lora_up, alpha, y);
}
